// MaskEstimator_80642305949763
// MI455X (gfx1250) — matrix operand order repaired and checked by reading, not compiled, not run on hardware
//
#include <hip/hip_runtime.h>
#include <hip/hip_bf16.h>

typedef __attribute__((ext_vector_type(16))) _Float16 v16h;
typedef __attribute__((ext_vector_type(8)))  _Float16 v8h;
typedef __attribute__((ext_vector_type(8)))  float    v8f;

#define NB   4
#define NC   256
#define NL   8000
#define NCR  64
#define NK   7
#define NBLK 24
#define EPSF 1e-5f

static_assert(NL % 32 == 0, "l-tiling assumes NL is a multiple of 32 (wave tile)");

// ---------------------------------------------------------------------------
// f32 -> f16 weight conversion
// ---------------------------------------------------------------------------
__global__ __launch_bounds__(256)
void f32_to_f16_kernel(const float* __restrict__ s, _Float16* __restrict__ d, int n) {
  int i = blockIdx.x * 256 + threadIdx.x;
  if (i < n) d[i] = (_Float16)s[i];
}

__global__ __launch_bounds__(256)
void zero_kernel(float* __restrict__ d, int n) {
  int i = blockIdx.x * 256 + threadIdx.x;
  if (i < n) d[i] = 0.f;
}

// ---------------------------------------------------------------------------
// Per-column (over C=256) mean / rstd for channel LayerNorm.
// One thread per (b,l) column; each strided read is coalesced across lanes.
// ---------------------------------------------------------------------------
__global__ __launch_bounds__(256)
void stats_kernel(const float* __restrict__ x, float* __restrict__ mean,
                  float* __restrict__ rstd) {
  const int b = blockIdx.y;
  const int l = blockIdx.x * 256 + threadIdx.x;
  if (l >= NL) return;
  const float* col = x + (size_t)b * NC * NL + l;
  float s = 0.f, s2 = 0.f;
  for (int c = 0; c < NC; ++c) {
    float v = col[(size_t)c * NL];
    s += v; s2 += v * v;
  }
  const float m  = s * (1.f / NC);
  const float vr = s2 * (1.f / NC) - m * m;
  mean[(size_t)b * NL + l] = m;
  rstd[(size_t)b * NL + l] = rsqrtf(vr + EPSF);
}

// ---------------------------------------------------------------------------
// kern[b,k,l] = sum_r wspan[k,r] * red[b,r,l]   (7x64 tiny GEMM, VALU)
// ---------------------------------------------------------------------------
__global__ __launch_bounds__(256)
void kern_kernel(const float* __restrict__ red, const float* __restrict__ wspan,
                 float* __restrict__ kern) {
  const int b = blockIdx.y;
  const int l = blockIdx.x * 256 + threadIdx.x;
  if (l >= NL) return;
  const float* rcol = red + (size_t)b * NCR * NL + l;
  float acc[NK];
#pragma unroll
  for (int k = 0; k < NK; ++k) acc[k] = 0.f;
  for (int r = 0; r < NCR; ++r) {
    const float rv = rcol[(size_t)r * NL];
#pragma unroll
    for (int k = 0; k < NK; ++k) acc[k] += wspan[k * NCR + r] * rv;
  }
#pragma unroll
  for (int k = 0; k < NK; ++k)
    kern[((size_t)b * NK + k) * NL + l] = acc[k];
}

// ---------------------------------------------------------------------------
// Involution gather + PReLU:
//   inv[b,c,l] = prelu( sum_k out[b,c,l+(k-3)*dil] * kern[b,k,l] , alpha )
// ---------------------------------------------------------------------------
__global__ __launch_bounds__(256)
void inv_kernel(const float* __restrict__ outb, const float* __restrict__ kern,
                float* __restrict__ inv, int dil, const float* __restrict__ alpha_p) {
  const int b = blockIdx.z;
  const int c = blockIdx.y;
  const int l = blockIdx.x * 256 + threadIdx.x;
  if (l >= NL) return;
  const float alpha = alpha_p[0];
  const float* oc = outb + ((size_t)b * NC + c) * NL;
  const float* kc = kern + (size_t)b * NK * NL + l;
  float acc = 0.f;
#pragma unroll
  for (int k = 0; k < NK; ++k) {
    const int pos = l + (k - 3) * dil;
    const float xv = (pos >= 0 && pos < NL) ? oc[pos] : 0.f;
    acc += xv * kc[(size_t)k * NL];
  }
  acc = (acc >= 0.f) ? acc : alpha * acc;
  inv[((size_t)b * NC + c) * NL + l] = acc;
}

// ---------------------------------------------------------------------------
// PReLU on skip accumulator (post block)
// ---------------------------------------------------------------------------
__global__ __launch_bounds__(256)
void prelu_kernel(const float* __restrict__ s, const float* __restrict__ alpha_p,
                  float* __restrict__ d, int n) {
  int i = blockIdx.x * 256 + threadIdx.x;
  if (i >= n) return;
  const float a = alpha_p[0];
  const float v = s[i];
  d[i] = (v >= 0.f) ? v : a * v;
}

// ---------------------------------------------------------------------------
// WMMA GEMM:  D[b,m,l] = sum_k A[m,k] * B_op(X[b,k,l])   (K = 256 fixed)
//   NORM: B_op(x) = (x - mean[b,l]) * rstd[b,l] * gam[k] + bet[k]   (fused LN)
//   RES : D += res   (res may alias D; per-element read-then-write per thread)
//   RELU: D = max(D, 0)
// Wave tile 32(m) x 32(l): 4 accumulators, 4 v_wmma per K-step.
// Block = 256 threads = 8 waves -> 64(m) x 128(l) tile.
//
// Bounds handling: NL % 32 == 0 and every wave's 32-column span is aligned,
// so a wave is either fully in-bounds or fully out-of-bounds. OOB waves exit
// up front (no barriers in this kernel); in-bounds waves run with ZERO
// predication: all loads/stores are unconditional -> no exec-mask traffic.
//
// A/B fragment layouts used here (contiguous k, same order for A and B):
//   A 16x32: lane m=lane&15;  halves [e]=k(16*hi + e)
//   B 32x16: lane n=lane&15;  halves [e]=k(16*hi + e)
//   C/D f32: lane n=lane&15;  vgpr r -> m = r + 8*hi
// ---------------------------------------------------------------------------
template<bool NORM, bool RELU, bool RES>
__global__ __launch_bounds__(256)
void wmma_gemm_kernel(const _Float16* __restrict__ A,   // M x 256 row-major (f16)
                      const float* __restrict__ X,      // NB x 256 x NL (f32)
                      const float* __restrict__ mean,   // NB x NL   (NORM)
                      const float* __restrict__ rstd,   // NB x NL   (NORM)
                      const float* __restrict__ gam,    // 256       (NORM)
                      const float* __restrict__ bet,    // 256       (NORM)
                      const float* res,                 // NB x M x NL (RES)
                      float* D,                         // NB x M x NL
                      int M) {
  constexpr int K = 256;
  const int b    = blockIdx.z;
  const int lane = threadIdx.x & 31;
  const int wv   = threadIdx.x >> 5;
  const int ml   = lane & 15;
  const int hi   = lane >> 4;
  const int m0   = blockIdx.y * 64 + (wv >> 2) * 32;
  const int l0   = blockIdx.x * 128 + (wv & 3) * 32;

  // Wave-uniform early exit: this wave's 32 columns are entirely OOB.
  if (l0 >= NL) return;

  const float* Xb = X + (size_t)b * K * NL;
  const int lc0 = l0 + ml;        // columns for B fragment 0 (fully in-bounds)
  const int lc1 = l0 + 16 + ml;   // columns for B fragment 1 (fully in-bounds)

  float mu0 = 0.f, rs0 = 1.f, mu1 = 0.f, rs1 = 1.f;
  if (NORM) {
    const float* mb = mean + (size_t)b * NL;
    const float* rb = rstd + (size_t)b * NL;
    mu0 = mb[lc0]; rs0 = rb[lc0];
    mu1 = mb[lc1]; rs1 = rb[lc1];
  }

  const _Float16* Ar0 = A + (size_t)(m0 + ml) * K;       // m-subtile 0
  const _Float16* Ar1 = A + (size_t)(m0 + 16 + ml) * K;  // m-subtile 1
  const float* Xc0 = Xb + lc0;
  const float* Xc1 = Xb + lc1;
  const int ka = hi * 16;   // A k-group offset for this half-wave
  const int kb = hi * 16;   // B k-group offset for this half-wave

  v8f acc00 = {}, acc01 = {}, acc10 = {}, acc11 = {};

#pragma unroll 2
  for (int k0 = 0; k0 < K; k0 += 32) {
    // A fragments: two 16B loads each (contiguous in k)
    v8h a0lo = *(const v8h*)(Ar0 + k0 + ka);
    v8h a0hi = *(const v8h*)(Ar0 + k0 + 8 + ka);
    v8h a1lo = *(const v8h*)(Ar1 + k0 + ka);
    v8h a1hi = *(const v8h*)(Ar1 + k0 + 8 + ka);
    v16h a0 = __builtin_shufflevector(a0lo, a0hi, 0,1,2,3,4,5,6,7,8,9,10,11,12,13,14,15);
    v16h a1 = __builtin_shufflevector(a1lo, a1hi, 0,1,2,3,4,5,6,7,8,9,10,11,12,13,14,15);

    // B fragments: built in-register from f32 activations with fused LN;
    // each element-load is coalesced across the wave (lanes = consecutive l)
    // and unconditional (no exec-mask manipulation).
    v16h b0, b1;
#pragma unroll
    for (int e = 0; e < 16; ++e) {
      const int k = k0 + kb + e;
      float x0 = Xc0[(size_t)k * NL];
      float x1 = Xc1[(size_t)k * NL];
      if (NORM) {
        const float g  = gam[k];
        const float be = bet[k];
        x0 = (x0 - mu0) * rs0 * g + be;
        x1 = (x1 - mu1) * rs1 * g + be;
      }
      b0[e] = (_Float16)x0;
      b1[e] = (_Float16)x1;
    }

    acc00 = __builtin_amdgcn_wmma_f32_16x16x32_f16(false, a0, false, b0, (short)0, acc00, false, false);
    acc01 = __builtin_amdgcn_wmma_f32_16x16x32_f16(false, a0, false, b1, (short)0, acc01, false, false);
    acc10 = __builtin_amdgcn_wmma_f32_16x16x32_f16(false, a1, false, b0, (short)0, acc10, false, false);
    acc11 = __builtin_amdgcn_wmma_f32_16x16x32_f16(false, a1, false, b1, (short)0, acc11, false, false);
  }

  // Store (optionally +residual, optionally ReLU) — unconditional.
#pragma unroll
  for (int i = 0; i < 2; ++i) {
#pragma unroll
    for (int j = 0; j < 2; ++j) {
      const int lc = (j == 0) ? lc0 : lc1;
      v8f av = (i == 0) ? ((j == 0) ? acc00 : acc01)
                        : ((j == 0) ? acc10 : acc11);
#pragma unroll
      for (int r = 0; r < 8; ++r) {
        const int    mr  = m0 + i * 16 + hi * 8 + r;
        const size_t off = ((size_t)b * M + mr) * NL + lc;
        float v = av[r];
        if (RES)  v += res[off];
        if (RELU) v = fmaxf(v, 0.f);
        D[off] = v;
      }
    }
  }
}

// ---------------------------------------------------------------------------
extern "C" void kernel_launch(void* const* d_in, const int* in_sizes, int n_in,
                              void* d_out, int out_size, void* d_ws, size_t ws_size,
                              hipStream_t stream) {
  const float* x       = (const float*)d_in[0];
  const float* iln_g   = (const float*)d_in[1];
  const float* iln_b   = (const float*)d_in[2];
  const float* iconv   = (const float*)d_in[3];
  const float* wreduce = (const float*)d_in[4];
  const float* wspan   = (const float*)d_in[5];
  const float* prelua  = (const float*)d_in[6];
  const float* lng     = (const float*)d_in[7];
  const float* lnb     = (const float*)d_in[8];
  const float* wmain   = (const float*)d_in[9];
  const float* wskip   = (const float*)d_in[10];
  const float* pprelua = (const float*)d_in[11];
  const float* wpost   = (const float*)d_in[12];
  float* out_final = (float*)d_out;
  (void)in_sizes; (void)n_in; (void)out_size; (void)ws_size;

  char* ws = (char*)d_ws;
  const size_t LB = (size_t)NB * NC * NL;   // 8,192,000 elements
  float* outb  = (float*)(ws);
  float* skip  = (float*)(ws + LB * 4);
  float* hbuf  = (float*)(ws + 2 * LB * 4);
  float* redb  = (float*)(ws + 3 * LB * 4);
  float* kernb = (float*)(ws + 3 * LB * 4 + (size_t)NB * NCR * NL * 4);
  float* meanb = (float*)(ws + 3 * LB * 4 + (size_t)NB * NCR * NL * 4 + (size_t)NB * NK * NL * 4);
  float* rstdb = meanb + (size_t)NB * NL;
  _Float16* wi_h  = (_Float16*)(rstdb + (size_t)NB * NL);
  _Float16* wr_h  = wi_h  + (size_t)NC * NC;
  _Float16* wm_h  = wr_h  + (size_t)NBLK * NCR * NC;
  _Float16* wsk_h = wm_h  + (size_t)NBLK * NC * NC;
  _Float16* wp_h  = wsk_h + (size_t)NBLK * NC * NC;

  const int T = 256;
  auto cdiv = [](int a, int b) { return (a + b - 1) / b; };
  const int GX = cdiv(NL, 128);   // 63 l-tiles (partial waves self-exit)

  // One-time f32 -> f16 weight conversion (all GEMM A-operands)
  f32_to_f16_kernel<<<cdiv(NC * NC, T), T, 0, stream>>>(iconv, wi_h, NC * NC);
  f32_to_f16_kernel<<<cdiv(NBLK * NCR * NC, T), T, 0, stream>>>(wreduce, wr_h, NBLK * NCR * NC);
  f32_to_f16_kernel<<<cdiv(NBLK * NC * NC, T), T, 0, stream>>>(wmain, wm_h, NBLK * NC * NC);
  f32_to_f16_kernel<<<cdiv(NBLK * NC * NC, T), T, 0, stream>>>(wskip, wsk_h, NBLK * NC * NC);
  f32_to_f16_kernel<<<cdiv(NC * NC, T), T, 0, stream>>>(wpost, wp_h, NC * NC);

  zero_kernel<<<cdiv((int)LB, T), T, 0, stream>>>(skip, (int)LB);

  // init: out = init_conv @ LN(x)   (LN fused into GEMM B-operand)
  stats_kernel<<<dim3(cdiv(NL, T), NB), T, 0, stream>>>(x, meanb, rstdb);
  wmma_gemm_kernel<true, false, false><<<dim3(GX, NC / 64, NB), T, 0, stream>>>(
      wi_h, x, meanb, rstdb, iln_g, iln_b, nullptr, outb, NC);

  for (int i = 0; i < NBLK; ++i) {
    const int dil = 1 << (i % 8);
    // red = relu(Wr @ out)
    wmma_gemm_kernel<false, true, false><<<dim3(GX, NCR / 64, NB), T, 0, stream>>>(
        wr_h + (size_t)i * NCR * NC, outb,
        nullptr, nullptr, nullptr, nullptr, nullptr, redb, NCR);
    // kern = Ws @ red
    kern_kernel<<<dim3(cdiv(NL, T), NB), T, 0, stream>>>(
        redb, wspan + (size_t)i * NK * NCR, kernb);
    // inv = prelu(involution(out, kern))
    inv_kernel<<<dim3(cdiv(NL, T), NC, NB), T, 0, stream>>>(
        outb, kernb, hbuf, dil, prelua + i);
    // LN stats over channels of inv
    stats_kernel<<<dim3(cdiv(NL, T), NB), T, 0, stream>>>(hbuf, meanb, rstdb);
    // skip += Wsk @ LN(inv)
    wmma_gemm_kernel<true, false, true><<<dim3(GX, NC / 64, NB), T, 0, stream>>>(
        wsk_h + (size_t)i * NC * NC, hbuf, meanb, rstdb,
        lng + (size_t)i * NC, lnb + (size_t)i * NC, skip, skip, NC);
    // out = Wm @ LN(inv) + out   (in-place residual, element-local RMW)
    wmma_gemm_kernel<true, false, true><<<dim3(GX, NC / 64, NB), T, 0, stream>>>(
        wm_h + (size_t)i * NC * NC, hbuf, meanb, rstdb,
        lng + (size_t)i * NC, lnb + (size_t)i * NC, outb, outb, NC);
  }

  // post: out_final = relu(Wp @ prelu(skip))
  prelu_kernel<<<cdiv((int)LB, T), T, 0, stream>>>(skip, pprelua, hbuf, (int)LB);
  wmma_gemm_kernel<false, true, false><<<dim3(GX, NC / 64, NB), T, 0, stream>>>(
      wp_h, hbuf, nullptr, nullptr, nullptr, nullptr, nullptr, out_final, NC);
}
